// AttnSageConv_4234837754498
// MI455X (gfx1250) — compile-verified, matrix operand order checked by reading, not run on hardware
//
#include <hip/hip_runtime.h>
#include <hip/hip_bf16.h>

// ---------------------------------------------------------------------------
// AttnSageConv for MI455X (gfx1250, wave32, WMMA).
//
// All dense math is [Mx128]@[128x128]-class GEMMs via v_wmma_f32_16x16x32_f16
// (K/V projected once per node instead of per gathered copy: 16x fewer FLOPs,
// mathematically identical). Attention softmax over <=17 keys is VALU.
// GEMM is fully templated (K, N, RMS, SILU, HASBIAS) so every instantiation
// is branch-free with fully unrolled fragment loads:
//   - A and B tiles staged to LDS as f16, B stored TRANSPOSED (Bs[n][k]) so
//     both fragments load as two ds_load_b128 per wmma.
//   - +8-half row padding => 4-bank stride per lane, conflict-free b128 loads.
//   - bias/silu/rmsnorm fused; silu uses v_rcp_f32 (no IEEE div sequence).
// ---------------------------------------------------------------------------

#define D    128
#define DH   32

typedef __attribute__((ext_vector_type(16))) _Float16 v16h;
typedef __attribute__((ext_vector_type(8)))  _Float16 v8h;
typedef __attribute__((ext_vector_type(8)))  float    v8f;

__device__ __forceinline__ float silu_f(float x) {
  return x * __builtin_amdgcn_rcpf(1.0f + __expf(-x));
}

// Y[M,N] = act( (RMS ? rmsnorm(A)*rmsW : A) @ W[K,N] + (HASB ? bias : 0) )
// M % 16 == 0. Block = 256 threads = 8 waves; 16-row tile per block; one
// 16-col tile per wave.
template <int K, int N, bool RMS, bool SILU, bool HASB>
__global__ __launch_bounds__(256) void gemm_wmma_kernel(
    const float* __restrict__ A, const float* __restrict__ W,
    const float* __restrict__ bias, float* __restrict__ Y,
    const float* __restrict__ rmsW)
{
  constexpr int AK  = K + 8;          // padded row stride (halfs): 4-bank skew
  constexpr int PER = K / 16;         // A elements per staging thread

  __shared__ _Float16 As[16 * AK];
  __shared__ _Float16 Bs[N * AK];     // TRANSPOSED: Bs[n*AK + k]
  __shared__ float    red[256];
  __shared__ float    rowScale[16];

  const int tid  = threadIdx.x;
  const int row0 = blockIdx.x * 16;

  __builtin_prefetch(A + (size_t)row0 * K, 0, 1);   // global_prefetch_b8

  // Stage W (row-major K x N in memory) transposed into LDS as f16.
  for (int i = tid; i < K * N; i += 256) {
    const int k = i / N;
    const int n = i - k * N;
    Bs[n * AK + k] = (_Float16)W[i];
  }

  // Stage A tile (16 rows x K) with optional fused RMSNorm.
  {
    const int r  = tid >> 4;          // row in tile 0..15
    const int c0 = tid & 15;          // col lane 0..15
    const float* Arow = A + (size_t)(row0 + r) * K;
    float vals[PER];
    float ss = 0.0f;
#pragma unroll
    for (int i = 0; i < PER; ++i) {
      const float v = Arow[c0 + i * 16];
      vals[i] = v;
      ss += v * v;
    }
    if (RMS) {
      red[tid] = ss;
      __syncthreads();
      if (c0 == 0) {
        float s = 0.0f;
#pragma unroll
        for (int i = 0; i < 16; ++i) s += red[(r << 4) + i];
        rowScale[r] = rsqrtf(s / (float)K + 1e-6f);
      }
      __syncthreads();
      const float sc = rowScale[r];
#pragma unroll
      for (int i = 0; i < PER; ++i) {
        const int c = c0 + i * 16;
        As[r * AK + c] = (_Float16)(vals[i] * sc * rmsW[c]);
      }
    } else {
#pragma unroll
      for (int i = 0; i < PER; ++i) {
        As[r * AK + c0 + i * 16] = (_Float16)vals[i];
      }
    }
  }
  __syncthreads();

  const int wave = tid >> 5;
  const int lane = tid & 31;
  const int g    = lane >> 4;     // half-wave: K sub-bank / output-row offset
  const int ln   = lane & 15;     // A: row M; B,D: col N

  for (int t = wave; t < (N >> 4); t += 8) {
    v8f acc = {};
#pragma unroll
    for (int kb = 0; kb < (K >> 5); ++kb) {
      // Fragment K packing per ISA (16-bit A/B, 16x32):
      //   e<8  -> k = kb*32 + g*8 + e
      //   e>=8 -> k = kb*32 + 16 + g*8 + (e-8)
      const v8h alo = *(const v8h*)&As[ln * AK + kb * 32 + g * 8];
      const v8h ahi = *(const v8h*)&As[ln * AK + kb * 32 + 16 + g * 8];
      const int bn  = (t * 16 + ln) * AK + kb * 32 + g * 8;
      const v8h blo = *(const v8h*)&Bs[bn];
      const v8h bhi = *(const v8h*)&Bs[bn + 16];
      v16h a, b;
#pragma unroll
      for (int e = 0; e < 8; ++e) {
        a[e] = alo[e]; a[e + 8] = ahi[e];
        b[e] = blo[e]; b[e + 8] = bhi[e];
      }
      acc = __builtin_amdgcn_wmma_f32_16x16x32_f16(
          /*neg_a=*/false, a, /*neg_b=*/false, b,
          /*c_mod=*/(short)0, acc, /*reuse_a=*/false, /*reuse_b=*/false);
    }
    // D layout: lane ln = column; vgpr r -> row m = r + 8*g.
    const int col = t * 16 + ln;
    const float bv = HASB ? bias[col] : 0.0f;
#pragma unroll
    for (int r = 0; r < 8; ++r) {
      float v = acc[r] + bv;
      if (SILU) v = silu_f(v);
      Y[(size_t)(row0 + r + g * 8) * N + col] = v;
    }
  }
}

// Attention with precomputed Q/K/V rows. One thread per (node, head).
// kv = (selfFirst ? [node] : []) ++ nbr[node, 0..nk)
__global__ __launch_bounds__(256) void attn_nbr_kernel(
    const float* __restrict__ Q, const float* __restrict__ Kb,
    const float* __restrict__ Vb, const int* __restrict__ nbr,
    int nk, int selfFirst, int M, float* __restrict__ ctx)
{
  const int gid  = blockIdx.x * blockDim.x + threadIdx.x;
  const int node = gid >> 2;
  const int h    = gid & 3;
  if (node >= M) return;
  const float scale = 0.17677669529663687f;  // 1/sqrt(32)

  float qv[DH];
  const float* q = Q + (size_t)node * D + h * DH;
#pragma unroll
  for (int d = 0; d < DH; ++d) qv[d] = q[d];

  int idx[17];
  const int nkv = nk + selfFirst;
  if (selfFirst) idx[0] = node;
  for (int j = 0; j < nk; ++j) idx[j + selfFirst] = nbr[(size_t)node * nk + j];

  float s[17];
  float mx = -3.4e38f;
  for (int j = 0; j < nkv; ++j) {
    const float* kr = Kb + (size_t)idx[j] * D + h * DH;
    float dot = 0.0f;
#pragma unroll
    for (int d = 0; d < DH; ++d) dot += qv[d] * kr[d];
    s[j] = dot * scale;
    mx = fmaxf(mx, s[j]);
  }
  float denom = 0.0f;
  for (int j = 0; j < nkv; ++j) { s[j] = __expf(s[j] - mx); denom += s[j]; }
  const float inv = __builtin_amdgcn_rcpf(denom);

  float cv[DH];
#pragma unroll
  for (int d = 0; d < DH; ++d) cv[d] = 0.0f;
  for (int j = 0; j < nkv; ++j) {
    const float p = s[j] * inv;
    const float* vr = Vb + (size_t)idx[j] * D + h * DH;
#pragma unroll
    for (int d = 0; d < DH; ++d) cv[d] += p * vr[d];
  }
  float* o = ctx + (size_t)node * D + h * DH;
#pragma unroll
  for (int d = 0; d < DH; ++d) o[d] = cv[d];
}

// cell_attn: kv = [self (cell K/V)] ++ 16 mapped AIG K/V rows, ALiBi bias
// from topo: rel_j = (max_topo+1) - topo_j ; bias = rel>8 ? -1e9 : -0.1(h+1)rel.
__global__ __launch_bounds__(256) void attn_cell_kernel(
    const float* __restrict__ Q, const float* __restrict__ Ks,
    const float* __restrict__ Vs, const float* __restrict__ Kn,
    const float* __restrict__ Vn, const int* __restrict__ nbr,
    const float* __restrict__ topo, int M, float* __restrict__ ctx)
{
  const int gid  = blockIdx.x * blockDim.x + threadIdx.x;
  const int node = gid >> 2;
  const int h    = gid & 3;
  if (node >= M) return;
  const float scale = 0.17677669529663687f;
  const float slope = 0.1f * (float)(h + 1);

  float qv[DH];
  const float* q = Q + (size_t)node * D + h * DH;
#pragma unroll
  for (int d = 0; d < DH; ++d) qv[d] = q[d];

  int   idx[16];
  float tp[16];
  float mxt = -3.4e38f;
#pragma unroll
  for (int j = 0; j < 16; ++j) {
    idx[j] = nbr[(size_t)node * 16 + j];
    tp[j]  = topo[idx[j]];          // topo_aig is (N_AIG, 1)
    mxt = fmaxf(mxt, tp[j]);
  }
  const float maxp1 = mxt + 1.0f;

  float s[17];
  {
    const float* kr = Ks + (size_t)node * D + h * DH;
    float dot = 0.0f;
#pragma unroll
    for (int d = 0; d < DH; ++d) dot += qv[d] * kr[d];
    s[0] = dot * scale;             // self: rel = 0 -> bias 0
  }
  float mx = s[0];
#pragma unroll
  for (int j = 0; j < 16; ++j) {
    const float* kr = Kn + (size_t)idx[j] * D + h * DH;
    float dot = 0.0f;
#pragma unroll
    for (int d = 0; d < DH; ++d) dot += qv[d] * kr[d];
    const float rel = maxp1 - tp[j];
    const float b   = (rel > 8.0f) ? -1.0e9f : (-slope * rel);
    s[1 + j] = dot * scale + b;
    mx = fmaxf(mx, s[1 + j]);
  }
  float denom = 0.0f;
#pragma unroll
  for (int j = 0; j < 17; ++j) { s[j] = __expf(s[j] - mx); denom += s[j]; }
  const float inv = __builtin_amdgcn_rcpf(denom);

  float cv[DH];
#pragma unroll
  for (int d = 0; d < DH; ++d) cv[d] = s[0] * inv * Vs[(size_t)node * D + h * DH + d];
#pragma unroll
  for (int j = 0; j < 16; ++j) {
    const float p = s[1 + j] * inv;
    const float* vr = Vn + (size_t)idx[j] * D + h * DH;
#pragma unroll
    for (int d = 0; d < DH; ++d) cv[d] += p * vr[d];
  }
  float* o = ctx + (size_t)node * D + h * DH;
#pragma unroll
  for (int d = 0; d < DH; ++d) o[d] = cv[d];
}

// neigh_aig = (gate==1 ? o_and : o_not) + h_aig
__global__ void select_add_kernel(const float* __restrict__ oa,
                                  const float* __restrict__ on,
                                  const float* __restrict__ h,
                                  const int* __restrict__ gate,
                                  float* __restrict__ out, int M)
{
  const int i = blockIdx.x * blockDim.x + threadIdx.x;
  if (i >= M * D) return;
  const int node = i / D;
  const float v = (gate[node] == 1) ? oa[i] : on[i];
  out[i] = v + h[i];
}

// cat[:, 0:128] = o_ca + h_cell ; cat[:, 128:256] = o_cc + h_cell
__global__ void concat_add_kernel(const float* __restrict__ oca,
                                  const float* __restrict__ occ,
                                  const float* __restrict__ hc,
                                  float* __restrict__ cat, int M)
{
  const int i = blockIdx.x * blockDim.x + threadIdx.x;
  if (i >= M * D) return;
  const int node = i / D;
  const int c    = i - node * D;
  const float hs = hc[i];
  cat[(size_t)node * 256 + c]       = oca[i] + hs;
  cat[(size_t)node * 256 + 128 + c] = occ[i] + hs;
}

__global__ void rst_kernel(const float* __restrict__ sf,
                           const float* __restrict__ nf,
                           float* __restrict__ rst, int n)
{
  const int i = blockIdx.x * blockDim.x + threadIdx.x;
  if (i < n) rst[i] = silu_f(sf[i] + nf[i]);
}

__global__ void final_kernel(const float* __restrict__ m3,
                             const float* __restrict__ rst,
                             float* __restrict__ out, int n)
{
  const int i = blockIdx.x * blockDim.x + threadIdx.x;
  if (i < n) out[i] = silu_f(m3[i]) + rst[i];
}

// ---------------------------------------------------------------------------
// Input index map. Top-level inputs in setup_inputs() insertion order; the
// nested `params` pytree is flattened JAX-style (sorted dict keys, None bias
// of proj_and dropped): attn dicts -> k,o,q,v ; each _lin -> b,w.
// ---------------------------------------------------------------------------
enum {
  IN_FEAT_AIG = 0, IN_FEAT_CELL = 1, IN_TOPO = 2,
  P_AFTER_NORM = 3,
  P_AND_KB = 4,  P_AND_KW = 5,  P_AND_OB = 6,  P_AND_OW = 7,
  P_AND_QB = 8,  P_AND_QW = 9,  P_AND_VB = 10, P_AND_VW = 11,
  P_CC_KB = 12,  P_CC_KW = 13,  P_CC_OB = 14,  P_CC_OW = 15,
  P_CC_QB = 16,  P_CC_QW = 17,  P_CC_VB = 18,  P_CC_VW = 19,
  P_CA_KB = 20,  P_CA_KW = 21,  P_CA_OB = 22,  P_CA_OW = 23,
  P_CA_QB = 24,  P_CA_QW = 25,  P_CA_VB = 26,  P_CA_VW = 27,
  P_CF0_B = 28,  P_CF0_W = 29,  P_CF1_B = 30,  P_CF1_W = 31,
  P_M0_B = 32,   P_M0_W = 33,   P_M1_B = 34,   P_M1_W = 35,
  P_M2_B = 36,   P_M2_W = 37,
  P_NF_B = 38,   P_NF_W = 39,
  P_NOT_KB = 40, P_NOT_KW = 41, P_NOT_OB = 42, P_NOT_OW = 43,
  P_NOT_QB = 44, P_NOT_QW = 45, P_NOT_VB = 46, P_NOT_VW = 47,
  P_PRE_NORM = 48, P_PA_W = 49, P_PC_B = 50, P_PC_W = 51,
  P_SF_B = 52,   P_SF_W = 53,
  IN_GATE = 54, IN_NBR_AND = 55, IN_NBR_MAP = 56, IN_NBR_CC = 57
};

extern "C" void kernel_launch(void* const* d_in, const int* in_sizes, int n_in,
                              void* d_out, int out_size, void* d_ws, size_t ws_size,
                              hipStream_t stream) {
  (void)in_sizes; (void)n_in; (void)out_size; (void)ws_size;
  const size_t NA = 40000, NC = 20000, NT = 60000;

  auto fin = [&](int i) { return (const float*)d_in[i]; };

  const float* feat_aig  = fin(IN_FEAT_AIG);
  const float* feat_cell = fin(IN_FEAT_CELL);
  const float* topo      = fin(IN_TOPO);
  const int*   gate      = (const int*)d_in[IN_GATE];
  const int*   nbr_and   = (const int*)d_in[IN_NBR_AND];
  const int*   nbr_map   = (const int*)d_in[IN_NBR_MAP];
  const int*   nbr_cc    = (const int*)d_in[IN_NBR_CC];

  // ---- workspace layout (floats), with lifetime-safe aliasing -------------
  float* ws = (float*)d_ws;
  size_t off = 0;
  auto take = [&](size_t n) { float* p = ws + off; off += n; return p; };

  float* h_all  = take(NT * D);          // h_aig ++ h_cell (contiguous)
  float* h_aig  = h_all;
  float* h_cell = h_all + NA * D;

  float* qa    = take(NA * D);
  float* ka    = take(NA * D);
  float* va    = take(NA * D);
  float* qn    = take(NA * D);
  float* kn    = take(NA * D);
  float* vn    = take(NA * D);
  float* qca   = take(NC * D);
  float* kca_a = take(NA * D);
  float* vca_a = take(NA * D);
  float* kca_c = take(NC * D);
  float* vca_c = take(NC * D);
  float* qcc   = take(NC * D);
  float* kcc   = take(NC * D);
  float* vcc   = take(NC * D);

  float* ctx_and = take(NA * D);
  float* ctx_not = take(NA * D);
  float* ctx_ca  = take(NC * D);
  float* ctx_cc  = take(NC * D);
  float* neigh   = take(NT * D);

  // Aliases over buffers that are dead by the time these are written.
  // (GEMM in-place A==Y is safe: each block stages its 16 A rows to LDS
  //  before writing the same 16 Y rows; no cross-block row sharing.)
  float* o_and   = ctx_and;
  float* o_not   = ctx_not;
  float* o_ca    = ctx_ca;
  float* o_cc    = ctx_cc;
  float* cat     = qa;      // NC*256 == NA*D
  float* cross_h = ka;      // NC*32
  float* sf      = va;      // NT*D spans va..mid-qn (dead)
  float* nf      = kn;      // NT*D spans kn..mid-vn (dead)
  float* rstb    = qca;     // NT*D spans qca+kca_a (dead)
  float* m1b     = vca_a;   // NT*32
  float* m2b     = vca_a + NT * 32;
  float* m3b     = kca_c;   // NT*D spans kca_c,vca_c,qcc (dead)

  const float* pre_norm   = fin(P_PRE_NORM);
  const float* after_norm = fin(P_AFTER_NORM);

#define G128(A_, W_, B_, Y_, M_) \
  gemm_wmma_kernel<128, 128, false, false, true> \
      <<<dim3((unsigned)((M_) / 16)), 256, 0, stream>>>(A_, W_, B_, Y_, nullptr)

  // 1) input projections (fused RMSNorm)
  gemm_wmma_kernel<128, 128, true, false, false>
      <<<dim3((unsigned)(NA / 16)), 256, 0, stream>>>(
          feat_aig, fin(P_PA_W), nullptr, h_aig, pre_norm);
  gemm_wmma_kernel<128, 128, true, false, true>
      <<<dim3((unsigned)(NC / 16)), 256, 0, stream>>>(
          feat_cell, fin(P_PC_W), fin(P_PC_B), h_cell, pre_norm);

  // 2) Q/K/V projections (project once per node, gather projected rows)
  G128(h_aig,  fin(P_AND_QW), fin(P_AND_QB), qa,    NA);
  G128(h_aig,  fin(P_AND_KW), fin(P_AND_KB), ka,    NA);
  G128(h_aig,  fin(P_AND_VW), fin(P_AND_VB), va,    NA);
  G128(h_aig,  fin(P_NOT_QW), fin(P_NOT_QB), qn,    NA);
  G128(h_aig,  fin(P_NOT_KW), fin(P_NOT_KB), kn,    NA);
  G128(h_aig,  fin(P_NOT_VW), fin(P_NOT_VB), vn,    NA);
  G128(h_cell, fin(P_CA_QW),  fin(P_CA_QB),  qca,   NC);
  G128(h_aig,  fin(P_CA_KW),  fin(P_CA_KB),  kca_a, NA);
  G128(h_aig,  fin(P_CA_VW),  fin(P_CA_VB),  vca_a, NA);
  G128(h_cell, fin(P_CA_KW),  fin(P_CA_KB),  kca_c, NC);
  G128(h_cell, fin(P_CA_VW),  fin(P_CA_VB),  vca_c, NC);
  G128(h_cell, fin(P_CC_QW),  fin(P_CC_QB),  qcc,   NC);
  G128(h_cell, fin(P_CC_KW),  fin(P_CC_KB),  kcc,   NC);
  G128(h_cell, fin(P_CC_VW),  fin(P_CC_VB),  vcc,   NC);

  // 3) attention cores (VALU, 1 thread per node*head)
  attn_nbr_kernel<<<dim3((unsigned)(NA * 4 / 256)), 256, 0, stream>>>(
      qa, ka, va, nbr_and, 16, 0, (int)NA, ctx_and);
  attn_nbr_kernel<<<dim3((unsigned)(NA * 4 / 256)), 256, 0, stream>>>(
      qn, kn, vn, nbr_and, 16, 0, (int)NA, ctx_not);
  attn_cell_kernel<<<dim3((unsigned)((NC * 4 + 255) / 256)), 256, 0, stream>>>(
      qca, kca_c, vca_c, kca_a, vca_a, nbr_map, topo, (int)NC, ctx_ca);
  attn_nbr_kernel<<<dim3((unsigned)((NC * 4 + 255) / 256)), 256, 0, stream>>>(
      qcc, kcc, vcc, nbr_cc, 8, /*selfFirst=*/1, (int)NC, ctx_cc);

  // 4) O projections (in-place over ctx, safe per block-row mapping)
  G128(ctx_and, fin(P_AND_OW), fin(P_AND_OB), o_and, NA);
  G128(ctx_not, fin(P_NOT_OW), fin(P_NOT_OB), o_not, NA);
  G128(ctx_ca,  fin(P_CA_OW),  fin(P_CA_OB),  o_ca,  NC);
  G128(ctx_cc,  fin(P_CC_OW),  fin(P_CC_OB),  o_cc,  NC);

  // 5) combine: neigh_aig = select + self; cell: concat -> cross_fc MLP
  select_add_kernel<<<dim3((unsigned)(NA * D / 256)), 256, 0, stream>>>(
      o_and, o_not, h_aig, gate, neigh, (int)NA);
  concat_add_kernel<<<dim3((unsigned)(NC * D / 256)), 256, 0, stream>>>(
      o_ca, o_cc, h_cell, cat, (int)NC);
  gemm_wmma_kernel<256, 32, false, true, true>
      <<<dim3((unsigned)(NC / 16)), 256, 0, stream>>>(
          cat, fin(P_CF0_W), fin(P_CF0_B), cross_h, nullptr);
  gemm_wmma_kernel<32, 128, false, false, true>
      <<<dim3((unsigned)(NC / 16)), 256, 0, stream>>>(
          cross_h, fin(P_CF1_W), fin(P_CF1_B), neigh + NA * D, nullptr);

  // 6) post block on all 60000 rows (aig ++ cell, matching d_out order)
  G128(h_all, fin(P_SF_W), fin(P_SF_B), sf, NT);
  G128(neigh, fin(P_NF_W), fin(P_NF_B), nf, NT);
  rst_kernel<<<dim3((unsigned)(NT * D / 256)), 256, 0, stream>>>(
      sf, nf, rstb, (int)(NT * D));
  gemm_wmma_kernel<128, 32, true, true, true>
      <<<dim3((unsigned)(NT / 16)), 256, 0, stream>>>(
          rstb, fin(P_M0_W), fin(P_M0_B), m1b, after_norm);
  gemm_wmma_kernel<32, 32, false, true, true>
      <<<dim3((unsigned)(NT / 16)), 256, 0, stream>>>(
          m1b, fin(P_M1_W), fin(P_M1_B), m2b, nullptr);
  gemm_wmma_kernel<32, 128, false, false, true>
      <<<dim3((unsigned)(NT / 16)), 256, 0, stream>>>(
          m2b, fin(P_M2_W), fin(P_M2_B), m3b, nullptr);
  final_kernel<<<dim3((unsigned)(NT * D / 256)), 256, 0, stream>>>(
      m3b, rstb, (float*)d_out, (int)(NT * D));
#undef G128
}
